// MIL_8615704395807
// MI455X (gfx1250) — matrix operand order repaired and checked by reading, not compiled, not run on hardware
//
#include <hip/hip_runtime.h>
#include <hip/hip_bf16.h>
#include <math.h>

// ---------------------------------------------------------------------------
// Types for WMMA
// ---------------------------------------------------------------------------
typedef __attribute__((ext_vector_type(16))) __bf16 v16bf;
typedef __attribute__((ext_vector_type(8)))  __bf16 v8bf;
typedef __attribute__((ext_vector_type(8)))  float  v8f;

#define HDIM   512
#define CATD   1536
#define DLAST  4608

__device__ __forceinline__ unsigned short f2bf(float f) {
    union { float f; unsigned u; } x; x.f = f;
    unsigned u = x.u;
    u += 0x7fffu + ((u >> 16) & 1u);       // round-to-nearest-even
    return (unsigned short)(u >> 16);
}

// K-swizzle within each aligned group of 32 (A-fragment-ready layout):
// groups of 8 reordered 0,2,1,3 so one lane's 16 values are contiguous.
__device__ __forceinline__ int swzl(int c /*0..31*/) {
    const int g  = c >> 3;
    const int g2 = ((g & 1) << 1) | (g >> 1);
    return (c & 7) + (g2 << 3);
}
__device__ __forceinline__ int swz32(int k) {
    return (k & ~31) + swzl(k & 31);
}

// ---------------------------------------------------------------------------
// gfx1250 async global->LDS copy (ASYNCcnt) + waits
// ---------------------------------------------------------------------------
__device__ __forceinline__ void async_b128(unsigned short* lds_ptr, const void* g) {
    unsigned lds_off = (unsigned)(unsigned long long)lds_ptr; // low 32b of flat = LDS offset
    asm volatile("global_load_async_to_lds_b128 %0, %1, off"
                 :: "v"(lds_off), "v"(g) : "memory");
}
__device__ __forceinline__ void wait_async0() {
    asm volatile("s_wait_asynccnt 0x0" ::: "memory");
}
__device__ __forceinline__ void wait_async4() {
    asm volatile("s_wait_asynccnt 0x4" ::: "memory");  // oldest stage (4 ops) done
}

// ---------------------------------------------------------------------------
// Tiled bf16 WMMA GEMM on pre-converted operands, double-buffered async LDS.
//   A  : [M x K] bf16, K-swizzled (fragment-ready)
//   Bt : [N x K] bf16, transposed, K-swizzled like A (fragment-ready)
// Block tile 64x64, BK=32, 128 threads = 4 waves, wave = 32x32 (2x2 WMMA).
// Requires M%64==0, N%64==0, K%32==0 (true for every call here).
// ACT: 0=none 1=leaky 2=tanh 3=sigmoid.  WF32: write f32 C.  WBF: write
// bf16-swizzled copy Cb (fragment-ready for a downstream GEMM with K=N).
// ---------------------------------------------------------------------------
template<int ACT, int BIAS, int ACCUM, int MUL, int WF32, int WBF>
__global__ __launch_bounds__(128)
void gemm_bf16_wmma(const unsigned short* __restrict__ A,
                    const unsigned short* __restrict__ Bt,
                    const float* __restrict__ bias, const float* __restrict__ mul,
                    float* __restrict__ C, unsigned short* __restrict__ Cb,
                    int M, int N, int K)
{
    const int bm = blockIdx.x * 64;
    const int bn = blockIdx.y * 64;

    __shared__ __align__(16) unsigned short As[2][64 * 40]; // 32 bf16 + pad per row
    __shared__ __align__(16) unsigned short Bs[2][64 * 40];

    v8f acc[2][2] = {};

    const int tid  = threadIdx.x;
    const int lane = tid & 31;
    const int wv   = tid >> 5;
    const int wm   = (wv >> 1) * 32;
    const int wn   = (wv & 1) * 32;
    const int half = lane >> 4;
    const int l    = lane & 15;

    const int r  = tid >> 1;          // row of the 64-row tile this thread stages
    const int kh = (tid & 1) * 16;    // which 16-element half of BK=32

    const unsigned short* ga0 = A  + (size_t)(bm + r) * K + kh;
    const unsigned short* gb0 = Bt + (size_t)(bn + r) * K + kh;
    const int lofs = r * 40 + kh;

    const int nk = K >> 5;

    // prologue: stage 0 into buffer 0
    {
        async_b128(&As[0][lofs],     ga0);
        async_b128(&As[0][lofs + 8], ga0 + 8);
        async_b128(&Bs[0][lofs],     gb0);
        async_b128(&Bs[0][lofs + 8], gb0 + 8);
    }

    for (int i = 0; i < nk; ++i) {
        // issue next stage into the other buffer, then wait only for the old one
        if (i + 1 < nk) {
            const int nb = (i + 1) & 1;
            const unsigned short* ga = ga0 + (size_t)(i + 1) * 32;
            const unsigned short* gb = gb0 + (size_t)(i + 1) * 32;
            async_b128(&As[nb][lofs],     ga);
            async_b128(&As[nb][lofs + 8], ga + 8);
            async_b128(&Bs[nb][lofs],     gb);
            async_b128(&Bs[nb][lofs + 8], gb + 8);
            wait_async4();                 // oldest 4 (stage i) have landed
        } else {
            wait_async0();
        }
        __syncthreads();

        // ---- load fragments (contiguous 32B per lane) and issue WMMAs ----
        const int cb = i & 1;
        v16bf af[2], bfr[2];
        #pragma unroll
        for (int t = 0; t < 2; ++t) {
            const v8bf* pa = (const v8bf*)(&As[cb][(wm + t * 16 + l) * 40 + half * 16]);
            const v8bf* pb = (const v8bf*)(&Bs[cb][(wn + t * 16 + l) * 40 + half * 16]);
            const v8bf alo = pa[0], ahi = pa[1];
            const v8bf blo = pb[0], bhi = pb[1];
            #pragma unroll
            for (int q = 0; q < 8; ++q) {
                af[t][q] = alo[q];  af[t][q + 8] = ahi[q];
                bfr[t][q] = blo[q]; bfr[t][q + 8] = bhi[q];
            }
        }
        #pragma unroll
        for (int ti = 0; ti < 2; ++ti)
            #pragma unroll
            for (int tj = 0; tj < 2; ++tj)
                acc[ti][tj] = __builtin_amdgcn_wmma_f32_16x16x32_bf16(
                    false, af[ti], false, bfr[tj], (short)0, acc[ti][tj], false, false);
        __syncthreads();   // all waves done reading buf cb before it is refilled
    }

    // ---- epilogue ----
    #pragma unroll
    for (int ti = 0; ti < 2; ++ti) {
        #pragma unroll
        for (int tj = 0; tj < 2; ++tj) {
            const int col  = bn + wn + tj * 16 + l;
            const int colS = swz32(col);      // bf16-copy position (K-swizzle of col)
            #pragma unroll
            for (int rr = 0; rr < 8; ++rr) {
                const int row = bm + wm + ti * 16 + half * 8 + rr;
                const size_t idx = (size_t)row * N + col;
                float v = acc[ti][tj][rr];
                if (ACCUM) v += C[idx];
                if (BIAS)  v += bias[col];
                if (ACT == 1) v = (v >= 0.f) ? v : 0.01f * v;
                else if (ACT == 2) v = tanhf(v);
                else if (ACT == 3) v = 1.f / (1.f + expf(-v));
                if (MUL) v *= mul[idx];
                if (WF32) C[idx] = v;
                if (WBF)  Cb[(size_t)row * N + colS] = f2bf(v);
            }
        }
    }
}

// ---------------------------------------------------------------------------
// Conversion kernels (run once per launch; bandwidth-trivial)
// ---------------------------------------------------------------------------
// f32 [rows x K] -> bf16 K-swizzled (A-operand layout)
__global__ void cvt_swz_kernel(const float* __restrict__ src,
                               unsigned short* __restrict__ dst,
                               size_t total, int K) {
    size_t i = (size_t)blockIdx.x * blockDim.x + threadIdx.x;
    if (i >= total) return;
    const int k = (int)(i % (size_t)K);
    dst[i - (k & 31) + swzl(k & 31)] = f2bf(src[i]);
}

// f32 W[K x N] -> bf16 Bt[N x K] (B-operand layout, K-swizzled)
__global__ void cvt_tr_kernel(const float* __restrict__ W,
                              unsigned short* __restrict__ Bt,
                              int K, int N) {
    size_t i = (size_t)blockIdx.x * blockDim.x + threadIdx.x;
    if (i >= (size_t)K * N) return;
    const int k = (int)(i / (size_t)N);
    const int n = (int)(i % (size_t)N);
    Bt[(size_t)n * K + swz32(k)] = f2bf(W[i]);
}

// ---------------------------------------------------------------------------
// Helper kernels
// ---------------------------------------------------------------------------
__global__ void zero_kernel(float* p, size_t n) {
    size_t i = (size_t)blockIdx.x * blockDim.x + threadIdx.x;
    if (i < n) p[i] = 0.f;
}

// agg[dst-rm] += x[src]   (H=512, float4 per thread, 128 threads per edge)
__global__ void seg_sum_kernel(const float* __restrict__ x, const int* __restrict__ src,
                               const int* __restrict__ dst, float* __restrict__ agg,
                               int E, int rm) {
    size_t t = (size_t)blockIdx.x * blockDim.x + threadIdx.x;
    int e  = (int)(t >> 7);
    int c4 = (int)(t & 127) * 4;
    if (e >= E) return;
    int s = src[e];
    int d = dst[e] - rm;
    const float4 v = *(const float4*)(x + (size_t)s * HDIM + c4);
    float* o = agg + (size_t)d * HDIM + c4;
    atomicAdd(o + 0, v.x); atomicAdd(o + 1, v.y);
    atomicAdd(o + 2, v.z); atomicAdd(o + 3, v.w);
}

// per-row layernorm over 512 + leaky; writes f32 x slice, bf16-swizzled x1b
// slice, and (optionally) the bf16-swizzled cats column block.
__global__ __launch_bounds__(128)
void ln_leaky_kernel(const float* __restrict__ h, const float* __restrict__ g,
                     const float* __restrict__ b, float* __restrict__ xout,
                     unsigned short* __restrict__ xbout,
                     unsigned short* __restrict__ coutb, int sz) {
    const int row = blockIdx.x;
    const int tid = threadIdx.x;
    __shared__ float r1[128], r2[128];
    float v[4]; float s = 0.f, s2 = 0.f;
    const float* hp = h + (size_t)row * HDIM;
    #pragma unroll
    for (int q = 0; q < 4; ++q) {
        v[q] = hp[tid * 4 + q];
        s += v[q]; s2 += v[q] * v[q];
    }
    r1[tid] = s; r2[tid] = s2; __syncthreads();
    for (int st = 64; st > 0; st >>= 1) {
        if (tid < st) { r1[tid] += r1[tid + st]; r2[tid] += r2[tid + st]; }
        __syncthreads();
    }
    const float mean = r1[0] * (1.f / HDIM);
    const float var  = r2[0] * (1.f / HDIM) - mean * mean;
    const float rstd = rsqrtf(var + 1e-5f);
    float* xo = xout + (size_t)row * HDIM;
    unsigned short* xb = xbout + (size_t)row * HDIM;
    unsigned short* cb = coutb ? coutb + (size_t)row * CATD : nullptr;
    #pragma unroll
    for (int q = 0; q < 4; ++q) {
        const int c = tid * 4 + q;
        float o = (v[q] - mean) * rstd * g[c] + b[c];
        o = (o >= 0.f) ? o : 0.01f * o;
        const int cs = swz32(c);
        const unsigned short ob = f2bf(o);
        xo[c] = o;
        xb[cs] = ob;
        if (cb) cb[cs] = ob;
    }
}

// catsb[:,0:512] = x1b rows (both already bf16-swizzled; straight copy)
__global__ void copy_cats_kernel(const unsigned short* __restrict__ xb,
                                 unsigned short* __restrict__ catsb, int sz) {
    size_t i = (size_t)blockIdx.x * blockDim.x + threadIdx.x;
    if (i >= (size_t)sz * HDIM) return;
    size_t row = i >> 9;
    int p = (int)(i & 511);
    catsb[row * CATD + p] = xb[i];
}

// out[row] = dot(prod[row,:1536], w) + b3[0]
__global__ __launch_bounds__(128)
void row_dot_kernel(const float* __restrict__ m, const float* __restrict__ w,
                    const float* __restrict__ b3, float* __restrict__ out) {
    const int row = blockIdx.x;
    const int tid = threadIdx.x;
    __shared__ float red[128];
    float s = 0.f;
    const float* mp = m + (size_t)row * CATD;
    for (int j = tid; j < CATD; j += 128) s += mp[j] * w[j];
    red[tid] = s; __syncthreads();
    for (int st = 64; st > 0; st >>= 1) {
        if (tid < st) red[tid] += red[tid + st];
        __syncthreads();
    }
    if (tid == 0) out[row] = red[0] + b3[0];
}

// single-block softmax over n; writes atts output and normalized buffer
__global__ __launch_bounds__(1024)
void softmax_kernel(const float* __restrict__ s, int n,
                    float* __restrict__ outAtt, float* __restrict__ aN) {
    const int tid = threadIdx.x;
    __shared__ float red[1024];
    float m = -1e30f;
    for (int i = tid; i < n; i += 1024) m = fmaxf(m, s[i]);
    red[tid] = m; __syncthreads();
    for (int st = 512; st > 0; st >>= 1) {
        if (tid < st) red[tid] = fmaxf(red[tid], red[tid + st]);
        __syncthreads();
    }
    const float mx = red[0]; __syncthreads();
    float acc = 0.f;
    for (int i = tid; i < n; i += 1024) acc += expf(s[i] - mx);
    red[tid] = acc; __syncthreads();
    for (int st = 512; st > 0; st >>= 1) {
        if (tid < st) red[tid] += red[tid + st];
        __syncthreads();
    }
    const float inv = 1.f / red[0];
    for (int i = tid; i < n; i += 1024) {
        float v = expf(s[i] - mx) * inv;
        outAtt[i] = v;
        aN[i] = v;
    }
}

// xv[c] = sum_r a[r]*xs[r][c]  over 1536 columns
__global__ __launch_bounds__(256)
void wsum_kernel(const float* __restrict__ a, const float* __restrict__ xs,
                 float* __restrict__ xv, int sz) {
    const int c = blockIdx.x * 256 + threadIdx.x;
    if (c >= CATD) return;
    float acc = 0.f;
    for (int r = 0; r < sz; ++r) acc += a[r] * xs[(size_t)r * CATD + c];
    xv[c] = acc;
}

// out[n] = leaky(sum_k v[k]*W[k,n] + b[n])
__global__ __launch_bounds__(256)
void dense_vec_kernel(const float* __restrict__ v, const float* __restrict__ W,
                      const float* __restrict__ b, float* __restrict__ out,
                      int Din, int Dout) {
    const int n = blockIdx.x * 256 + threadIdx.x;
    if (n >= Dout) return;
    float s = 0.f;
    for (int k = 0; k < Din; ++k) s += v[k] * W[(size_t)k * Dout + n];
    s += b[n];
    out[n] = (s >= 0.f) ? s : 0.01f * s;
}

// pred = sigmoid(t @ cla_w + cla_b), 2 outputs
__global__ __launch_bounds__(256)
void cla_kernel(const float* __restrict__ t, const float* __restrict__ W,
                const float* __restrict__ b, float* __restrict__ out) {
    const int tid = threadIdx.x;
    __shared__ float r0[256], r1[256];
    float s0 = 0.f, s1 = 0.f;
    for (int k = tid; k < DLAST; k += 256) {
        const float tv = t[k];
        s0 += tv * W[(size_t)k * 2 + 0];
        s1 += tv * W[(size_t)k * 2 + 1];
    }
    r0[tid] = s0; r1[tid] = s1; __syncthreads();
    for (int st = 128; st > 0; st >>= 1) {
        if (tid < st) { r0[tid] += r0[tid + st]; r1[tid] += r1[tid + st]; }
        __syncthreads();
    }
    if (tid == 0) {
        out[0] = 1.f / (1.f + expf(-(r0[0] + b[0])));
        out[1] = 1.f / (1.f + expf(-(r1[0] + b[1])));
    }
}

// ---------------------------------------------------------------------------
// Orchestration
// ---------------------------------------------------------------------------
static inline int gblk(size_t n) { return (int)((n + 255) / 256); }

extern "C" void kernel_launch(void* const* d_in, const int* in_sizes, int n_in,
                              void* d_out, int out_size, void* d_ws, size_t ws_size,
                              hipStream_t stream) {
    (void)in_sizes; (void)n_in; (void)out_size; (void)ws_size;

    static const int PS[3]  = {16000, 32000, 64000};
    static const int OFF[3] = {0, 16000, 48000};
    const int NTOT = 112000;

    const float* x_in     = (const float*)d_in[0];
    const int*   edges[3] = {(const int*)d_in[1], (const int*)d_in[2], (const int*)d_in[3]};
    const int*   diffs[2] = {(const int*)d_in[4], (const int*)d_in[5]};
    const float* w0     = (const float*)d_in[6];
    const float* b0     = (const float*)d_in[7];
    const float* cwrel  = (const float*)d_in[8];
    const float* cbrel  = (const float*)d_in[9];
    const float* cwroot = (const float*)d_in[10];
    const float* clng   = (const float*)d_in[11];
    const float* clnb   = (const float*)d_in[12];
    const float* dwrel  = (const float*)d_in[13];
    const float* dbrel  = (const float*)d_in[14];
    const float* dwroot = (const float*)d_in[15];
    const float* dlng   = (const float*)d_in[16];
    const float* dlnb   = (const float*)d_in[17];
    const float* attl1w = (const float*)d_in[18];
    const float* attl1b = (const float*)d_in[19];
    const float* att1w  = (const float*)d_in[20];
    const float* att1b  = (const float*)d_in[21];
    const float* att2w  = (const float*)d_in[22];
    const float* att2b  = (const float*)d_in[23];
    const float* att3w  = (const float*)d_in[24];
    const float* att3b  = (const float*)d_in[25];
    const float* lastw  = (const float*)d_in[26];
    const float* lastb  = (const float*)d_in[27];
    const float* claw   = (const float*)d_in[28];
    const float* clab   = (const float*)d_in[29];

    float* out = (float*)d_out;

    // ---- workspace layout: f32 region then bf16 (u16) region ----
    float* fw = (float*)d_ws;
    size_t o = 0;
    float* x1   = fw + o; o += (size_t)NTOT * HDIM;
    float* agg  = fw + o; o += (size_t)96000 * HDIM;
    float* hbuf = fw + o; o += (size_t)96000 * HDIM;
    float* xs   = fw + o; o += (size_t)64000 * CATD;
    float* a1b  = fw + o; o += (size_t)64000 * CATD;
    float* asc  = fw + o; o += (size_t)64000;
    float* xv   = fw + o; o += (size_t)DLAST;
    float* tbuf = fw + o; o += (size_t)DLAST;

    unsigned short* uw = (unsigned short*)(fw + o);
    size_t u = 0;
    unsigned short* xbf    = uw + u; u += (size_t)NTOT * 1024;   // input, swizzled
    unsigned short* x1b    = uw + u; u += (size_t)NTOT * HDIM;   // features, swizzled
    unsigned short* aggb   = uw + u; u += (size_t)96000 * HDIM;
    unsigned short* catsb  = uw + u; u += (size_t)NTOT * CATD;
    unsigned short* xsb    = uw + u; u += (size_t)64000 * CATD;
    unsigned short* w0t    = uw + u; u += (size_t)512 * 1024;    // [N x K]
    unsigned short* cwrelT = uw + u; u += (size_t)6 * HDIM * HDIM;
    unsigned short* cwrooT = uw + u; u += (size_t)6 * HDIM * HDIM;
    unsigned short* dwrelT = uw + u; u += (size_t)2 * HDIM * HDIM;
    unsigned short* dwrooT = uw + u; u += (size_t)2 * HDIM * HDIM;
    unsigned short* al1T   = uw + u; u += (size_t)3 * CATD * CATD;
    unsigned short* a1T    = uw + u; u += (size_t)3 * CATD * CATD;
    unsigned short* a2T    = uw + u; u += (size_t)3 * CATD * CATD;

    // ---- 0) one-time conversions ----
    cvt_swz_kernel<<<gblk((size_t)NTOT*1024), 256, 0, stream>>>(x_in, xbf, (size_t)NTOT*1024, 1024);
    cvt_tr_kernel<<<gblk((size_t)1024*512), 256, 0, stream>>>(w0, w0t, 1024, 512);
    for (int ij = 0; ij < 6; ++ij) {
        cvt_tr_kernel<<<gblk((size_t)HDIM*HDIM), 256, 0, stream>>>(
            cwrel + (size_t)ij*HDIM*HDIM, cwrelT + (size_t)ij*HDIM*HDIM, HDIM, HDIM);
        cvt_tr_kernel<<<gblk((size_t)HDIM*HDIM), 256, 0, stream>>>(
            cwroot + (size_t)ij*HDIM*HDIM, cwrooT + (size_t)ij*HDIM*HDIM, HDIM, HDIM);
    }
    for (int i = 0; i < 2; ++i) {
        cvt_tr_kernel<<<gblk((size_t)HDIM*HDIM), 256, 0, stream>>>(
            dwrel + (size_t)i*HDIM*HDIM, dwrelT + (size_t)i*HDIM*HDIM, HDIM, HDIM);
        cvt_tr_kernel<<<gblk((size_t)HDIM*HDIM), 256, 0, stream>>>(
            dwroot + (size_t)i*HDIM*HDIM, dwrooT + (size_t)i*HDIM*HDIM, HDIM, HDIM);
    }
    for (int i = 0; i < 3; ++i) {
        cvt_tr_kernel<<<gblk((size_t)CATD*CATD), 256, 0, stream>>>(
            attl1w + (size_t)i*CATD*CATD, al1T + (size_t)i*CATD*CATD, CATD, CATD);
        cvt_tr_kernel<<<gblk((size_t)CATD*CATD), 256, 0, stream>>>(
            att1w + (size_t)i*CATD*CATD, a1T + (size_t)i*CATD*CATD, CATD, CATD);
        cvt_tr_kernel<<<gblk((size_t)CATD*CATD), 256, 0, stream>>>(
            att2w + (size_t)i*CATD*CATD, a2T + (size_t)i*CATD*CATD, CATD, CATD);
    }

    // ---- 1) x1 = leaky(x @ w0 + b0); also write bf16 copy x1b ----
    gemm_bf16_wmma<1,1,0,0,1,1><<<dim3(NTOT/64, HDIM/64), 128, 0, stream>>>(
        xbf, w0t, b0, nullptr, x1, x1b, NTOT, HDIM, 1024);

    // ---- 2) per-segment GCN stack + diff layers ----
    for (int i = 0; i < 3; ++i) {
        const int rm = OFF[i], sz = PS[i], E = sz * 8;
        unsigned short* cats_i = catsb + (size_t)rm * CATD;

        copy_cats_kernel<<<gblk((size_t)sz*HDIM), 256, 0, stream>>>(
            x1b + (size_t)rm * HDIM, cats_i, sz);

        for (int j = 0; j < 2; ++j) {
            const int ij = i * 2 + j;
            zero_kernel<<<gblk((size_t)sz*HDIM), 256, 0, stream>>>(agg, (size_t)sz*HDIM);
            seg_sum_kernel<<<gblk((size_t)E*128), 256, 0, stream>>>(
                x1, edges[i], edges[i] + E, agg, E, rm);
            cvt_swz_kernel<<<gblk((size_t)sz*HDIM), 256, 0, stream>>>(
                agg, aggb, (size_t)sz*HDIM, HDIM);
            gemm_bf16_wmma<0,1,0,0,1,0><<<dim3(sz/64, HDIM/64), 128, 0, stream>>>(
                aggb, cwrelT + (size_t)ij*HDIM*HDIM, cbrel + (size_t)ij*HDIM, nullptr,
                hbuf, nullptr, sz, HDIM, HDIM);
            gemm_bf16_wmma<0,0,1,0,1,0><<<dim3(sz/64, HDIM/64), 128, 0, stream>>>(
                x1b + (size_t)rm*HDIM, cwrooT + (size_t)ij*HDIM*HDIM, nullptr, nullptr,
                hbuf, nullptr, sz, HDIM, HDIM);
            ln_leaky_kernel<<<sz, 128, 0, stream>>>(
                hbuf, clng + (size_t)ij*HDIM, clnb + (size_t)ij*HDIM,
                x1 + (size_t)rm*HDIM, x1b + (size_t)rm*HDIM,
                cats_i + (size_t)(j+1)*HDIM, sz);
        }

        if (i < 2) {
            const int sz2 = PS[i] + PS[i+1], E2 = sz2 * 8;
            zero_kernel<<<gblk((size_t)sz2*HDIM), 256, 0, stream>>>(agg, (size_t)sz2*HDIM);
            seg_sum_kernel<<<gblk((size_t)E2*128), 256, 0, stream>>>(
                x1, diffs[i], diffs[i] + E2, agg, E2, rm);
            cvt_swz_kernel<<<gblk((size_t)sz2*HDIM), 256, 0, stream>>>(
                agg, aggb, (size_t)sz2*HDIM, HDIM);
            gemm_bf16_wmma<0,1,0,0,1,0><<<dim3(sz2/64, HDIM/64), 128, 0, stream>>>(
                aggb, dwrelT + (size_t)i*HDIM*HDIM, dbrel + (size_t)i*HDIM, nullptr,
                hbuf, nullptr, sz2, HDIM, HDIM);
            gemm_bf16_wmma<0,0,1,0,1,0><<<dim3(sz2/64, HDIM/64), 128, 0, stream>>>(
                x1b + (size_t)rm*HDIM, dwrooT + (size_t)i*HDIM*HDIM, nullptr, nullptr,
                hbuf, nullptr, sz2, HDIM, HDIM);
            ln_leaky_kernel<<<sz2, 128, 0, stream>>>(
                hbuf, dlng + (size_t)i*HDIM, dlnb + (size_t)i*HDIM,
                x1 + (size_t)rm*HDIM, x1b + (size_t)rm*HDIM, nullptr, sz2);
        }
    }

    // ---- 3) attention pooling per segment ----
    const int attOut[3] = {2, 2 + 16000, 2 + 48000};
    for (int i = 0; i < 3; ++i) {
        const int sz = PS[i];
        unsigned short* cats_i = catsb + (size_t)OFF[i] * CATD;
        const size_t wq = (size_t)i * CATD * CATD;
        const size_t bq = (size_t)i * CATD;

        // xs = leaky(cats @ attl1_w + b); write f32 (for wsum) + bf16 (for att1/2)
        gemm_bf16_wmma<1,1,0,0,1,1><<<dim3(sz/64, CATD/64), 128, 0, stream>>>(
            cats_i, al1T + wq, attl1b + bq, nullptr, xs, xsb, sz, CATD, CATD);
        // a1 = tanh(xs @ att1_w + b)
        gemm_bf16_wmma<2,1,0,0,1,0><<<dim3(sz/64, CATD/64), 128, 0, stream>>>(
            xsb, a1T + wq, att1b + bq, nullptr, a1b, nullptr, sz, CATD, CATD);
        // prod = sigmoid(xs @ att2_w + b) * a1   (in place over a1b)
        gemm_bf16_wmma<3,1,0,1,1,0><<<dim3(sz/64, CATD/64), 128, 0, stream>>>(
            xsb, a2T + wq, att2b + bq, a1b, a1b, nullptr, sz, CATD, CATD);
        // scores, softmax, pooled vector
        row_dot_kernel<<<sz, 128, 0, stream>>>(a1b, att3w + bq, att3b + i, asc);
        softmax_kernel<<<1, 1024, 0, stream>>>(asc, sz, out + attOut[i], asc);
        wsum_kernel<<<CATD/256, 256, 0, stream>>>(asc, xs, xv + (size_t)i * CATD, sz);
    }

    // ---- 4) head ----
    dense_vec_kernel<<<(DLAST + 255)/256, 256, 0, stream>>>(xv, lastw, lastb, tbuf, DLAST, DLAST);
    cla_kernel<<<1, 256, 0, stream>>>(tbuf, claw, clab, out);
}
